// HEmbedding_15977278341160
// MI455X (gfx1250) — matrix operand order repaired and checked by reading, not compiled, not run on hardware
//
#include <hip/hip_runtime.h>
#include <hip/hip_bf16.h>

// ---------------------------------------------------------------------------
// Problem constants (from reference)
// ---------------------------------------------------------------------------
#define B_    128
#define L_    12
#define O_    128
#define MC_   512
#define NCV_  256
#define D_    512
#define FD_   4096
#define H1_   4096
#define NN_   (MC_ + O_)     // 640 nodes
#define HUGE_A 100.0f
#define TEMP_  10.0f
#define TH_    0.9f

typedef __attribute__((ext_vector_type(16))) __bf16         v16bf;
typedef __attribute__((ext_vector_type(8)))  float          v8f;
typedef __attribute__((ext_vector_type(4)))  unsigned int   u32x4;
typedef __attribute__((ext_vector_type(4)))  float          f32x4;
typedef __attribute__((ext_vector_type(4)))  unsigned short u16x4;

__device__ __forceinline__ float scale_att(float x) {
    return TEMP_ * (x - TH_) / (1.0f - TH_);
}

// fp32 -> bf16 (round to nearest even)
__device__ __forceinline__ unsigned short f2bf(float f) {
    unsigned int u = __float_as_uint(f);
    u += 0x7FFFu + ((u >> 16) & 1u);
    return (unsigned short)(u >> 16);
}

// ---------------------------------------------------------------------------
// Tiled bf16 WMMA GEMM:  Cout[M,N] = A[M,K] * B[K,N] (+ bias[N])
// Row-major fp32 in/out; operands converted to bf16 while staging into LDS.
// Block tile 128x128, 8 waves (wave32), each wave 32x64 = 2x4 WMMA tiles.
// Double-buffered LDS (2 x 16 KB) software pipeline:
//   iter k: [global_load_b128 tile k+1 -> regs] [WMMA from LDS buf]
//           [pack bf16 regs -> LDS buf^1] [barrier]
// Requires M%128==0, N%128==0, K%32==0 (true for all call sites).
// ---------------------------------------------------------------------------
#define BM 128
#define BN 128
#define BK 32

__global__ __launch_bounds__(256)
void gemm_bf16_wmma(const float* __restrict__ A, const float* __restrict__ Bm,
                    const float* __restrict__ bias, float* __restrict__ Cout,
                    int M, int N, int K)
{
    __shared__ __attribute__((aligned(16))) unsigned short As[2][BM * BK]; // [m][k]
    __shared__ __attribute__((aligned(16))) unsigned short Bs[2][BN * BK]; // [n][k]

    const int tid  = threadIdx.x;
    const int lane = tid & 31;
    const int wave = tid >> 5;          // 0..7
    const int l16  = lane & 15;
    const int half = lane >> 4;         // 0 or 1
    const int wy   = wave & 3;          // 4 waves down  (rows)
    const int wx   = wave >> 2;         // 2 waves across (cols)
    const int m0g  = blockIdx.y * BM;
    const int n0g  = blockIdx.x * BN;

    // staging indices (per thread: 4 x float4 for A, 4 x float4 for B)
    // A: e in [0,1024): r = e>>3 (row 0..127), c = (e&7)*4 (col 0..28)
    // B: e in [0,1024): kr = e>>5 (k 0..31),  nc = (e&31)*4 (n 0..124)
    f32x4 aR[4], bR[4];

    v8f acc[2][4];
    const v8f vz = {};
    for (int i = 0; i < 2; ++i)
        for (int j = 0; j < 4; ++j) acc[i][j] = vz;

    // ---- global load of one K-tile into registers (b128, coalesced) ----
    auto load_tile = [&](int k0) {
        #pragma unroll
        for (int j = 0; j < 4; ++j) {
            int e = tid + j * 256;
            int r = e >> 3, c = (e & 7) << 2;
            aR[j] = *(const f32x4*)&A[(size_t)(m0g + r) * K + (k0 + c)];
            int kr = e >> 5, nc = (e & 31) << 2;
            bR[j] = *(const f32x4*)&Bm[(size_t)(k0 + kr) * N + (n0g + nc)];
        }
    };
    // ---- pack registers to bf16 and store into LDS buffer ----
    auto store_tile = [&](int buf) {
        #pragma unroll
        for (int j = 0; j < 4; ++j) {
            int e = tid + j * 256;
            int r = e >> 3, c = (e & 7) << 2;
            u16x4 av = { f2bf(aR[j][0]), f2bf(aR[j][1]),
                         f2bf(aR[j][2]), f2bf(aR[j][3]) };
            *(u16x4*)&As[buf][r * BK + c] = av;          // 8B aligned store
            int kr = e >> 5, nc = (e & 31) << 2;
            #pragma unroll
            for (int i = 0; i < 4; ++i)                  // transpose to [n][k]
                Bs[buf][(nc + i) * BK + kr] = f2bf(bR[j][i]);
        }
    };

    // prologue: stage tile 0
    load_tile(0);
    store_tile(0);
    __syncthreads();

    int buf = 0;
    for (int k0 = 0; k0 < K; k0 += BK) {
        const bool has_next = (k0 + BK < K);
        if (has_next) load_tile(k0 + BK);   // VMEM in flight under the WMMAs

        // ---- fragments per ISA VGPR layouts ----
        // A 16x32 bf16: lane<16 holds M=l16, K={0..7,16..23}; lane>=16 K={8..15,24..31}
        v16bf afrag[2], bfrag[4];
        #pragma unroll
        for (int mt = 0; mt < 2; ++mt) {
            int row = wy * 32 + mt * 16 + l16;
            union { v16bf v; u32x4 q[2]; } u;
            u.q[0] = *(const u32x4*)&As[buf][row * BK + half * 8];
            u.q[1] = *(const u32x4*)&As[buf][row * BK + 16 + half * 8];
            afrag[mt] = u.v;
        }
        // B 32x16 bf16: lane holds N=l16, K={0..7,16..23} (lane<16); K={8..15,24..31} (lane>=16)
        #pragma unroll
        for (int nt = 0; nt < 4; ++nt) {
            int col = wx * 64 + nt * 16 + l16;
            union { v16bf v; u32x4 q[2]; } u;
            u.q[0] = *(const u32x4*)&Bs[buf][col * BK + half * 8];
            u.q[1] = *(const u32x4*)&Bs[buf][col * BK + 16 + half * 8];
            bfrag[nt] = u.v;
        }

        #pragma unroll
        for (int mt = 0; mt < 2; ++mt)
            #pragma unroll
            for (int nt = 0; nt < 4; ++nt)
                acc[mt][nt] = __builtin_amdgcn_wmma_f32_16x16x32_bf16(
                    false, afrag[mt], false, bfrag[nt],
                    (short)0, acc[mt][nt], false, false);

        if (has_next) {
            store_tile(buf ^ 1);            // write other buffer: no barrier needed first
            __syncthreads();                // publish buf^1; everyone done reading buf
            buf ^= 1;
        }
    }

    // ---- epilogue: C/D layout -> row = 8*half + i, col = l16 ----
    #pragma unroll
    for (int mt = 0; mt < 2; ++mt) {
        #pragma unroll
        for (int nt = 0; nt < 4; ++nt) {
            int col = n0g + wx * 64 + nt * 16 + l16;
            float bv = bias ? bias[col] : 0.0f;
            #pragma unroll
            for (int i = 0; i < 8; ++i) {
                int row = m0g + wy * 32 + mt * 16 + half * 8 + i;
                Cout[(size_t)row * N + col] = acc[mt][nt][i] + bv;
            }
        }
    }
}

// ---------------------------------------------------------------------------
// bias_eff[n] = b1 @ W2[:,n] + b2[n]   (n in [0,512))
// ---------------------------------------------------------------------------
__global__ void bias_eff_kernel(const float* __restrict__ b1,
                                const float* __restrict__ W2,
                                const float* __restrict__ b2,
                                float* __restrict__ be)
{
    int n = threadIdx.x;
    if (n >= D_) return;
    float s = b2[n];
    for (int k = 0; k < H1_; ++k) s += b1[k] * W2[(size_t)k * D_ + n];
    be[n] = s;
}

// ---------------------------------------------------------------------------
// In-place row L2-normalization, D == 512, one block (256 thr) per row
// ---------------------------------------------------------------------------
__global__ __launch_bounds__(256)
void row_normalize(float* __restrict__ X)
{
    __shared__ float red[256];
    const int row = blockIdx.x, tid = threadIdx.x;
    float a = X[(size_t)row * D_ + tid];
    float b = X[(size_t)row * D_ + tid + 256];
    red[tid] = a * a + b * b;
    __syncthreads();
    for (int s = 128; s > 0; s >>= 1) {
        if (tid < s) red[tid] += red[tid + s];
        __syncthreads();
    }
    float rn = rsqrtf(red[0]);
    X[(size_t)row * D_ + tid]       = a * rn;
    X[(size_t)row * D_ + tid + 256] = b * rn;
}

// ---------------------------------------------------------------------------
// Per-example program interpreter. One block (512 thr, 16 waves) per example.
// ---------------------------------------------------------------------------
__device__ __forceinline__ float blockReduceMax(float v, float* red) {
    int tid = threadIdx.x;
    red[tid] = v; __syncthreads();
    for (int s = 256; s > 0; s >>= 1) {
        if (tid < s) red[tid] = fmaxf(red[tid], red[tid + s]);
        __syncthreads();
    }
    float r = red[0]; __syncthreads();
    return r;
}
__device__ __forceinline__ float blockReduceSum(float v, float* red) {
    int tid = threadIdx.x;
    red[tid] = v; __syncthreads();
    for (int s = 256; s > 0; s >>= 1) {
        if (tid < s) red[tid] += red[tid + s];
        __syncthreads();
    }
    float r = red[0]; __syncthreads();
    return r;
}

__global__ __launch_bounds__(512)
void interp_kernel(const int* __restrict__ prog,      // [B,L,2]
                   const float* __restrict__ Cn,      // [512,512] unit rows
                   const float* __restrict__ objn,    // [B*128,512] unit rows
                   const float* __restrict__ relation_emb, // [512,512]
                   float* __restrict__ out)           // [B,256]
{
    __shared__ float att[NN_];
    __shared__ float chat[D_];
    __shared__ float red[512];

    const int b    = blockIdx.x;
    const int tid  = threadIdx.x;
    const int lane = tid & 31;
    const int wave = tid >> 5;   // 0..15
    const float* objb = objn + (size_t)b * O_ * D_;

    for (int n = tid; n < NN_; n += 512) att[n] = HUGE_A;
    __syncthreads();

    for (int l = 0; l < L_; ++l) {
        const int op  = prog[(b * L_ + l) * 2 + 0];
        const int arg = prog[(b * L_ + l) * 2 + 1];

        if (op == 1) {                                   // select
            const bool evenArg = ((arg & 1) == 0);
            for (int n = tid; n < NN_; n += 512) {
                bool cond = evenArg ? (n < MC_) : (n >= MC_);
                if (cond) att[n] = -HUGE_A;
            }
            __syncthreads();
        } else if (op == 2 || op == 3) {                 // filter / verify
            // chat = normalized concept row (Cn rows are already unit)
            chat[tid] = Cn[(size_t)arg * D_ + tid];
            __syncthreads();
            for (int n = wave; n < NN_; n += 16) {       // wave per node
                const float* row = (n < MC_) ? (Cn + (size_t)n * D_)
                                             : (objb + (size_t)(n - MC_) * D_);
                float p = 0.f;
                for (int d = lane; d < D_; d += 32) p += row[d] * chat[d];
                for (int off = 16; off >= 1; off >>= 1) p += __shfl_xor(p, off);
                if (lane == 0) {
                    float sc = scale_att(p);
                    if (op == 2) att[n] = fminf(att[n], sc);
                    else         att[n] = (n == arg) ? fminf(att[n], sc) : -HUGE_A;
                }
            }
            __syncthreads();
        } else if (op == 4) {                            // choose
            for (int n = tid; n < NN_; n += 512)
                if (n < MC_) att[n] = -HUGE_A;
            __syncthreads();
            if (tid == 0) att[arg] = HUGE_A;
            __syncthreads();
        } else if (op == 5) {                            // exist
            float v = -HUGE_A;
            for (int n = tid; n < NN_; n += 512) {
                float m = ((n >= NCV_) && (n < MC_)) ? -HUGE_A : att[n];
                v = fmaxf(v, m);
            }
            float yes = blockReduceMax(v, red);
            for (int n = tid; n < NN_; n += 512) att[n] = -HUGE_A;
            __syncthreads();
            if (tid == 0) { att[0] = yes; att[1] = -yes; }
            __syncthreads();
        } else if (op == 6) {                            // transfer
            float v = -1e30f;
            for (int n = tid; n < NN_; n += 512) v = fmaxf(v, att[n]);
            float m = blockReduceMax(v, red);
            float psum = 0.f;
            for (int n = tid; n < NN_; n += 512) {
                float e = expf(att[n] - m);
                att[n] = e;                              // att now holds exp weights
                psum += e;
            }
            __syncthreads();
            float s = blockReduceSum(psum, red);
            // gather g[d] = softmax(att) . nodes[:,d]   (d = tid, coalesced rows)
            float g = 0.f;
            for (int n = 0; n < NN_; ++n) {
                const float* row = (n < MC_) ? (Cn + (size_t)n * D_)
                                             : (objb + (size_t)(n - MC_) * D_);
                g += att[n] * row[tid];
            }
            g /= s;
            float t  = g + relation_emb[(size_t)arg * D_ + tid];
            float ss = blockReduceSum(t * t, red);       // syncs: g-phase done
            chat[tid] = t * rsqrtf(ss);
            __syncthreads();
            for (int n = wave; n < NN_; n += 16) {       // new att = scale(sims)
                const float* row = (n < MC_) ? (Cn + (size_t)n * D_)
                                             : (objb + (size_t)(n - MC_) * D_);
                float p = 0.f;
                for (int d = lane; d < D_; d += 32) p += row[d] * chat[d];
                for (int off = 16; off >= 1; off >>= 1) p += __shfl_xor(p, off);
                if (lane == 0) att[n] = scale_att(p);
            }
            __syncthreads();
        }
        // op == 0: null, nothing to do
    }

    // log_softmax over att[0:256]
    float v = (tid < NCV_) ? att[tid] : -1e30f;
    float m = blockReduceMax(v, red);
    float e = (tid < NCV_) ? expf(att[tid] - m) : 0.f;
    float s = blockReduceSum(e, red);
    if (tid < NCV_) out[(size_t)b * NCV_ + tid] = att[tid] - m - logf(s);
}

// ---------------------------------------------------------------------------
// Launcher
// ---------------------------------------------------------------------------
extern "C" void kernel_launch(void* const* d_in, const int* in_sizes, int n_in,
                              void* d_out, int out_size, void* d_ws, size_t ws_size,
                              hipStream_t stream)
{
    const float* scene         = (const float*)d_in[0];  // [128,128,4096]
    const int*   program       = (const int*)  d_in[1];  // [128,12,2]
    const float* concept_emb_h = (const float*)d_in[2];  // [512,512]
    const float* concept_W     = (const float*)d_in[3];  // [512,512]
    const float* concept_b     = (const float*)d_in[4];  // [512]
    const float* relation_emb  = (const float*)d_in[5];  // [512,512]
    const float* W1            = (const float*)d_in[6];  // [4096,4096]
    const float* b1            = (const float*)d_in[7];  // [4096]
    const float* W2            = (const float*)d_in[8];  // [4096,512]
    const float* b2            = (const float*)d_in[9];  // [512]
    float*       out           = (float*)d_out;          // [128,256]

    // Workspace layout (all fully written before read; ~41 MB total)
    char*  ws       = (char*)d_ws;
    float* Weff     = (float*)(ws);                        // 4096*512   =  8 MB
    float* Cmat     = (float*)(ws + 8388608);              // 512*512    =  1 MB
    float* bias_eff = (float*)(ws + 9437184);              // 512        =  2 KB
    float* obj      = (float*)(ws + 9439232);              // 16384*512  = 32 MB

    dim3 blk(256);

    // 1) Weff = W1 @ W2           (linear-layer fusion: no activation between)
    gemm_bf16_wmma<<<dim3(D_ / BN, H1_ / BM), blk, 0, stream>>>(
        W1, W2, nullptr, Weff, H1_, D_, FD_);

    // 2) bias_eff = b1 @ W2 + b2
    bias_eff_kernel<<<1, 512, 0, stream>>>(b1, W2, b2, bias_eff);

    // 3) obj = scene @ Weff + bias_eff   ([16384,4096] x [4096,512])
    gemm_bf16_wmma<<<dim3(D_ / BN, (B_ * O_) / BM), blk, 0, stream>>>(
        scene, Weff, bias_eff, obj, B_ * O_, D_, FD_);

    // 4) C = concept_emb_h @ concept_W + concept_b
    gemm_bf16_wmma<<<dim3(D_ / BN, MC_ / BM), blk, 0, stream>>>(
        concept_emb_h, concept_W, concept_b, Cmat, MC_, D_, 512);

    // 5) normalize node rows in place (Cn shared across batch; objn per object)
    row_normalize<<<MC_,      256, 0, stream>>>(Cmat);
    row_normalize<<<B_ * O_,  256, 0, stream>>>(obj);

    // 6) per-example interpreter + final log_softmax
    interp_kernel<<<B_, 512, 0, stream>>>(program, Cmat, obj, relation_emb, out);
}
